// BertCorrector_76742475645452
// MI455X (gfx1250) — matrix operand order repaired and checked by reading, not compiled, not run on hardware
//
#include <hip/hip_runtime.h>
#include <hip/hip_bf16.h>

// Problem sizes (fixed by the reference)
#define B_   32
#define S_   512
#define H_   768
#define V_   8192
#define WMAX 256
#define M_   (B_ * WMAX)   // 8192 GEMM rows

typedef __attribute__((ext_vector_type(16))) __bf16 v16bf;
typedef __attribute__((ext_vector_type(8)))  float  v8f;

// ---------------------------------------------------------------------------
// Kernel 1: segment mean.  merged[b*WMAX + w][h] = mean of enc rows whose
// (sorted) segment id == w, 0 if empty.  Output in bf16 for the WMMA GEMM.
// grid = B_*WMAX blocks, 256 threads.
// ---------------------------------------------------------------------------
__global__ __launch_bounds__(256) void seg_mean_kernel(
    const float* __restrict__ enc,      // [B, S, H] fp32
    const int*   __restrict__ seg,      // [B, S] sorted per row
    __bf16*      __restrict__ merged)   // [B*WMAX, H] bf16
{
    const int bw = blockIdx.x;
    const int b  = bw >> 8;             // / WMAX
    const int w  = bw & (WMAX - 1);
    const int* ids = seg + (size_t)b * S_;

    // lower_bound(w)
    int lo;
    {
        int l = 0, r = S_;
        while (l < r) { int m = (l + r) >> 1; if (ids[m] < w) l = m + 1; else r = m; }
        lo = l;
    }
    // upper_bound(w)
    int hi;
    {
        int l = lo, r = S_;
        while (l < r) { int m = (l + r) >> 1; if (ids[m] <= w) l = m + 1; else r = m; }
        hi = l;
    }
    const int   cnt = hi - lo;
    const float inv = (cnt > 0) ? (1.0f / (float)cnt) : 0.0f;

    const float* base = enc + ((size_t)b * S_ + (size_t)lo) * H_;
    __bf16* out = merged + (size_t)bw * H_;

    for (int h = threadIdx.x; h < H_; h += 256) {   // 3 iterations (768/256)
        float s = 0.0f;
        for (int t = 0; t < cnt; ++t) s += base[(size_t)t * H_ + h];
        out[h] = (__bf16)(s * inv);
    }
}

// ---------------------------------------------------------------------------
// Kernel 2: W [H, V] fp32 row-major  ->  Wt [V, H] bf16 (transposed), so the
// GEMM's B-matrix fragments are contiguous K-runs in LDS/global.
// grid = (V/32, H/32), block = (32, 8); LDS 32x33 tile transpose.
// ---------------------------------------------------------------------------
__global__ __launch_bounds__(256) void transpose_w_kernel(
    const float* __restrict__ W,        // [H, V] fp32
    __bf16*      __restrict__ Wt)       // [V, H] bf16
{
    __shared__ float tile[32][33];
    const int v0 = blockIdx.x * 32;
    const int h0 = blockIdx.y * 32;
    const int x  = threadIdx.x;         // 0..31
    const int y  = threadIdx.y;         // 0..7

    #pragma unroll
    for (int i = 0; i < 4; ++i) {
        const int h = h0 + y + i * 8;
        tile[y + i * 8][x] = W[(size_t)h * V_ + v0 + x];
    }
    __syncthreads();
    #pragma unroll
    for (int i = 0; i < 4; ++i) {
        const int v = v0 + y + i * 8;
        Wt[(size_t)v * H_ + h0 + x] = (__bf16)tile[x][y + i * 8];
    }
}

// ---------------------------------------------------------------------------
// Kernel 3: GEMM  C[m, n] = sum_k A[m,k] * Wt[n,k] + bias[n]
// Block tile 128x128, BK=32, 256 threads = 8 waves (wave32) as 2 (M) x 4 (N);
// each wave computes 4x2 subtiles of 16x16 via v_wmma_f32_16x16x32_bf16.
// Tiles staged with GLOBAL_LOAD_ASYNC_TO_LDS_B128 (ASYNCcnt) into a
// double-buffered LDS pipeline: wait/barrier, issue next tile, compute.
// ---------------------------------------------------------------------------
#define BM 128
#define BN 128
#define BK 32
#define LDS_STRIDE 40   // 32 + 8 pad (bf16 elements); rows stay 16B-aligned

// Issue one 32-byte chunk (two async b128 loads; offset: advances BOTH the
// LDS and global address per the TDM/async pseudocode).
#define ASYNC_LDG32(ldsaddr, voff, sbase)                                     \
    asm volatile("global_load_async_to_lds_b128 %0, %1, %2\n\t"               \
                 "global_load_async_to_lds_b128 %0, %1, %2 offset:16"         \
                 :: "v"(ldsaddr), "v"(voff), "s"(sbase) : "memory")

static __device__ inline v16bf load_frag2(const __bf16* p0, const __bf16* p1) {
    union { uint4 q[2]; v16bf v; } u;
    u.q[0] = *(const uint4*)p0;   // 8 contiguous bf16
    u.q[1] = *(const uint4*)p1;   // 8 contiguous bf16
    return u.v;
}

__global__ __launch_bounds__(256) void gemm_bf16_kernel(
    const __bf16* __restrict__ A,     // [M_, H_] merged, K contiguous
    const __bf16* __restrict__ Bt,    // [V_, H_] W transposed, K contiguous
    const float*  __restrict__ bias,  // [V_]
    float*        __restrict__ C)     // [M_, V_]
{
    __shared__ __bf16 sA[2][BM * LDS_STRIDE];
    __shared__ __bf16 sB[2][BN * LDS_STRIDE];

    const int tid  = threadIdx.x;
    const int wave = tid >> 5;
    const int lane = tid & 31;
    const int wm   = wave >> 2;          // 0..1 : 64 rows each
    const int wn   = wave & 3;           // 0..3 : 32 cols each
    const int lhi  = lane >> 4;          // lane half (0/1)
    const int lmod = lane & 15;

    const int m_block = blockIdx.y * BM;
    const int n_block = blockIdx.x * BN;

    // Tile staging: 256 threads x 32 B covers one 128x32 bf16 tile (8 KB).
    const int lrow  = tid >> 1;          // 0..127
    const int lhalf = (tid & 1) << 4;    // element offset 0 or 16

    // GVS addressing: saddr = uniform base, vaddr = per-thread 32-bit offset.
    const unsigned long long sbaseA = (unsigned long long)(A  + (size_t)m_block * H_);
    const unsigned long long sbaseB = (unsigned long long)(Bt + (size_t)n_block * H_);
    const unsigned voff0 = (unsigned)((lrow * H_ + lhalf) * sizeof(__bf16));

    const unsigned ldsrow = (unsigned)((lrow * LDS_STRIDE + lhalf) * sizeof(__bf16));
    const unsigned ldsA0 = (unsigned)(uintptr_t)(&sA[0][0]) + ldsrow;
    const unsigned ldsA1 = (unsigned)(uintptr_t)(&sA[1][0]) + ldsrow;
    const unsigned ldsB0 = (unsigned)(uintptr_t)(&sB[0][0]) + ldsrow;
    const unsigned ldsB1 = (unsigned)(uintptr_t)(&sB[1][0]) + ldsrow;

    v8f acc[4][2] = {};

    // Prologue: fetch K-tile 0 into buffer 0.
    ASYNC_LDG32(ldsA0, voff0, sbaseA);
    ASYNC_LDG32(ldsB0, voff0, sbaseB);

    const int NT = H_ / BK;              // 24 K-tiles
    for (int kt = 0; kt < NT; ++kt) {
        const int cur = kt & 1;

        // My async loads for buffer `cur` are done; barrier makes everyone's
        // visible (and guarantees the other buffer is no longer being read).
        asm volatile("s_wait_asynccnt 0x0" ::: "memory");
        __syncthreads();

        // Overlap: fetch K-tile kt+1 into the other buffer while we compute.
        if (kt + 1 < NT) {
            const unsigned voff = voff0 + (unsigned)((kt + 1) * BK * sizeof(__bf16));
            ASYNC_LDG32(cur ? ldsA0 : ldsA1, voff, sbaseA);
            ASYNC_LDG32(cur ? ldsB0 : ldsB1, voff, sbaseB);
        }

        const __bf16* bufA = sA[cur];
        const __bf16* bufB = sB[cur];

        // B fragments: per lane 16 contiguous K at offset lhi*16 of row N.
        v16bf bfrag[2];
        #pragma unroll
        for (int j = 0; j < 2; ++j) {
            const __bf16* p = bufB + (wn * 32 + j * 16 + lmod) * LDS_STRIDE + lhi * 16;
            bfrag[j] = load_frag2(p, p + 8);
        }
        // A fragments: per lane 16 contiguous K at offset lhi*16 of row M.
        #pragma unroll
        for (int i = 0; i < 4; ++i) {
            const __bf16* p = bufA + (wm * 64 + i * 16 + lmod) * LDS_STRIDE + lhi * 16;
            const v16bf afrag = load_frag2(p, p + 8);
            #pragma unroll
            for (int j = 0; j < 2; ++j) {
                acc[i][j] = __builtin_amdgcn_wmma_f32_16x16x32_bf16(
                    /*neg_a=*/false, afrag, /*neg_b=*/false, bfrag[j],
                    /*c_mod=*/(short)0, acc[i][j],
                    /*reuse_a=*/false, /*reuse_b=*/false);
            }
        }
    }

    // Epilogue: C/D layout => VGPR r holds (M = sub_m + 8*lhi + r, N = lmod).
    #pragma unroll
    for (int j = 0; j < 2; ++j) {
        const int n  = n_block + wn * 32 + j * 16 + lmod;
        const float bv = bias[n];
        #pragma unroll
        for (int i = 0; i < 4; ++i) {
            const int mbase = m_block + wm * 64 + i * 16 + lhi * 8;
            #pragma unroll
            for (int r = 0; r < 8; ++r) {
                C[(size_t)(mbase + r) * V_ + n] = acc[i][j][r] + bv;
            }
        }
    }
}

// ---------------------------------------------------------------------------
// Launch
// ---------------------------------------------------------------------------
extern "C" void kernel_launch(void* const* d_in, const int* in_sizes, int n_in,
                              void* d_out, int out_size, void* d_ws, size_t ws_size,
                              hipStream_t stream) {
    (void)in_sizes; (void)n_in; (void)out_size; (void)ws_size;

    const float* enc  = (const float*)d_in[0];   // [B, S, H]
    const int*   seg  = (const int*)  d_in[1];   // [B, S]
    const float* W    = (const float*)d_in[2];   // [H, V]
    const float* bias = (const float*)d_in[3];   // [V]
    // d_in[4] = num_words (== WMAX), compile-time constant here
    float* out = (float*)d_out;                  // [B, WMAX, V]

    // Workspace: merged bf16 [M_, H_] then Wt bf16 [V_, H_]
    __bf16* merged = (__bf16*)d_ws;
    __bf16* Wt     = (__bf16*)((char*)d_ws + (size_t)M_ * H_ * sizeof(__bf16));

    seg_mean_kernel<<<B_ * WMAX, 256, 0, stream>>>(enc, seg, merged);
    transpose_w_kernel<<<dim3(V_ / 32, H_ / 32), dim3(32, 8), 0, stream>>>(W, Wt);
    gemm_bf16_kernel<<<dim3(V_ / BN, M_ / BM), 256, 0, stream>>>(merged, Wt, bias, out);
}
